// NDTracker_62706522521640
// MI455X (gfx1250) — matrix operand order repaired and checked by reading, not compiled, not run on hardware
//
#include <hip/hip_runtime.h>
#include <hip/hip_bf16.h>
#include <math.h>

// ---------------- problem constants ----------------
constexpr int cB  = 4;
constexpr int cN  = 2048;
constexpr int cD  = 512;
constexpr int cH  = 8;
constexpr int cHD = 64;
constexpr int cFF = 1024;
constexpr int cC2 = 128;
constexpr int cHF = 64;
constexpr int cWF = 64;
constexpr int cM  = cB * cN;          // 8192 tokens
constexpr float cEPS = 1e-6f;

// ---------------- WMMA fragment types ----------------
typedef __attribute__((ext_vector_type(16))) _Float16 v16h;
typedef __attribute__((ext_vector_type(8)))  _Float16 v8h;
typedef __attribute__((ext_vector_type(8)))  float    v8f;

// A-matrix fragment (16x32 f16, row-major source, lda in elements).
// Per ISA 7.12.2: lane L holds M=L&15; v0..3 = K=(L>>4)*8+0..7, v4..7 = +16.
// Both halves are 16B-contiguous -> two b128 loads.
__device__ __forceinline__ v16h load_a_frag(const _Float16* A, int lda,
                                            int m0, int k0, int lane) {
  int m  = m0 + (lane & 15);
  int kb = k0 + ((lane >> 4) << 3);
  const v8h* p0 = (const v8h*)(A + (size_t)m * lda + kb);
  const v8h* p1 = (const v8h*)(A + (size_t)m * lda + kb + 16);
  v8h x0 = *p0, x1 = *p1;
  return __builtin_shufflevector(x0, x1, 0,1,2,3,4,5,6,7,8,9,10,11,12,13,14,15);
}

// B-matrix fragment (32x16 f16) sourced from a row-major (Nout,K) weight,
// i.e. B[k][n] = W[n][k]. Lane L holds N=L&15, K=(L>>4)*8+0..7, +16
// -> two 16B-contiguous halves = two b128 loads.
__device__ __forceinline__ v16h load_b_frag(const _Float16* W, int ldw,
                                            int n0, int k0, int lane) {
  int n  = n0 + (lane & 15);
  int kb = k0 + ((lane >> 4) << 3);
  const v8h* p = (const v8h*)(W + (size_t)n * ldw + kb);
  v8h x0 = p[0], x1 = p[2];
  return __builtin_shufflevector(x0, x1, 0,1,2,3,4,5,6,7,8,9,10,11,12,13,14,15);
}

__device__ __forceinline__ v8f wmma_f16(v16h a, v16h b, v8f c) {
  // (neg_a, A, neg_b, B, c_mod, C, reuse_a, reuse_b)
  return __builtin_amdgcn_wmma_f32_16x16x32_f16(false, a, false, b,
                                                (short)0, c, false, false);
}

__device__ __forceinline__ void zero4(v8f acc[4]) {
#pragma unroll
  for (int j = 0; j < 4; ++j)
#pragma unroll
    for (int r = 0; r < 8; ++r) acc[j][r] = 0.f;
}

// One wave accumulates a 32x64 output block (two 16-row subtiles sharing each
// B fragment): 8 independent WMMAs per 32-deep k-step -> latency hiding + 2x
// B-fragment reuse vs a 16x64 tile.
__device__ __forceinline__ void gemm_32x64(const _Float16* A, int lda,
                                           const _Float16* W, int ldw, int K,
                                           int m0, int n0, int lane,
                                           v8f acc0[4], v8f acc1[4]) {
  for (int k0 = 0; k0 < K; k0 += 32) {
    v16h a0 = load_a_frag(A, lda, m0,      k0, lane);
    v16h a1 = load_a_frag(A, lda, m0 + 16, k0, lane);
#pragma unroll
    for (int j = 0; j < 4; ++j) {
      v16h b = load_b_frag(W, ldw, n0 + 16 * j, k0, lane);
      acc0[j] = wmma_f16(a0, b, acc0[j]);
      acc1[j] = wmma_f16(a1, b, acc1[j]);
    }
  }
}

// C-layout epilogues. Lane holds (M = r + 8*(lane>>4), N = lane&15) in vgpr r.
__device__ __forceinline__ void store_resid16(v8f acc[4], const float* xin,
                                              float* xout, int m0, int n0, int lane) {
  int hi = lane >> 4, nc = lane & 15;
#pragma unroll
  for (int j = 0; j < 4; ++j) {
    int n = n0 + 16 * j + nc;
#pragma unroll
    for (int r = 0; r < 8; ++r) {
      size_t idx = (size_t)(m0 + r + 8 * hi) * cD + n;
      xout[idx] = xin[idx] + acc[j][r];
    }
  }
}

__device__ __forceinline__ void store_f16_16(v8f acc[4], _Float16* out, int ldo,
                                             int m0, int n0, int lane) {
  int hi = lane >> 4, nc = lane & 15;
#pragma unroll
  for (int j = 0; j < 4; ++j) {
    int n = n0 + 16 * j + nc;
#pragma unroll
    for (int r = 0; r < 8; ++r)
      out[(size_t)(m0 + r + 8 * hi) * ldo + n] = (_Float16)acc[j][r];
  }
}

// ---------------- elementwise / prep kernels ----------------
__global__ __launch_bounds__(256) void k_cvt_f16(const float* __restrict__ in,
                                                 _Float16* __restrict__ out, int n) {
  int i = blockIdx.x * blockDim.x + threadIdx.x;
  if (i < n) out[i] = (_Float16)in[i];
}

// Bilinear gather: one thread per (token, 4 channels).
__global__ __launch_bounds__(256) void k_feats(const float* __restrict__ fmaps,
                                               const float* __restrict__ coords,
                                               _Float16* __restrict__ feats) {
  int t = blockIdx.x * blockDim.x + threadIdx.x;
  int m = t >> 5;
  int c0 = (t & 31) << 2;
  if (m >= cM) return;
  int b = m / cN;
  float xc = fminf(fmaxf(coords[2 * m + 0], 0.f), (float)(cWF - 1));
  float yc = fminf(fmaxf(coords[2 * m + 1], 0.f), (float)(cHF - 1));
  float x0f = floorf(xc), y0f = floorf(yc);
  float wx = xc - x0f, wy = yc - y0f;
  int x0 = (int)x0f, y0 = (int)y0f;
  int x1 = min(x0 + 1, cWF - 1), y1 = min(y0 + 1, cHF - 1);
  const float* img = fmaps + (size_t)b * cC2 * cHF * cWF;
#pragma unroll
  for (int cc = 0; cc < 4; ++cc) {
    const float* ch = img + (size_t)(c0 + cc) * cHF * cWF;
    float f00 = ch[y0 * cWF + x0], f01 = ch[y0 * cWF + x1];
    float f10 = ch[y1 * cWF + x0], f11 = ch[y1 * cWF + x1];
    float top = f00 * (1.f - wx) + f01 * wx;
    float bot = f10 * (1.f - wx) + f11 * wx;
    feats[(size_t)m * cC2 + c0 + cc] = (_Float16)(top * (1.f - wy) + bot * wy);
  }
}

// Wave-per-row RMS norm, f32 in -> f16 out.
__global__ __launch_bounds__(256) void k_rms(const float* __restrict__ x,
                                             const float* __restrict__ scale,
                                             _Float16* __restrict__ out,
                                             int rows, int d) {
  int wave = (blockIdx.x * blockDim.x + threadIdx.x) >> 5;
  int lane = threadIdx.x & 31;
  if (wave >= rows) return;
  const float* r = x + (size_t)wave * d;
  float ss = 0.f;
  for (int i = lane; i < d; i += 32) { float v = r[i]; ss += v * v; }
#pragma unroll
  for (int o = 16; o > 0; o >>= 1) ss += __shfl_xor(ss, o, 32);
  float inv = rsqrtf(ss / (float)d + cEPS);
  _Float16* o = out + (size_t)wave * d;
  for (int i = lane; i < d; i += 32) o[i] = (_Float16)(r[i] * scale[i] * inv);
}

// ---------------- GEMM kernels (32x64 per wave) ----------------
// C = xin + A @ W^T, f32 output (ld = cD). A is (cM, K) f16, W is (cD, K) f16.
__global__ __launch_bounds__(256) void k_gemm_resid(const _Float16* __restrict__ A,
                                                    const _Float16* __restrict__ W,
                                                    const float* __restrict__ xin,
                                                    float* __restrict__ xout, int K) {
  int wave = (blockIdx.x * blockDim.x + threadIdx.x) >> 5;
  int lane = threadIdx.x & 31;
  const int MT = cM / 32;
  int mt = wave % MT, nt = wave / MT;
  if (nt >= cD / 64) return;
  int m0 = mt * 32, n0 = nt * 64;
  v8f acc0[4], acc1[4]; zero4(acc0); zero4(acc1);
  gemm_32x64(A, K, W, K, K, m0, n0, lane, acc0, acc1);
  store_resid16(acc0, xin, xout, m0,      n0, lane);
  store_resid16(acc1, xin, xout, m0 + 16, n0, lane);
}

// QKV: (cM, 3*cD) f16 output.
__global__ __launch_bounds__(256) void k_gemm_qkv(const _Float16* __restrict__ A,
                                                  const _Float16* __restrict__ W,
                                                  _Float16* __restrict__ qkv) {
  int wave = (blockIdx.x * blockDim.x + threadIdx.x) >> 5;
  int lane = threadIdx.x & 31;
  const int MT = cM / 32;
  int mt = wave % MT, nt = wave / MT;
  if (nt >= (3 * cD) / 64) return;
  int m0 = mt * 32, n0 = nt * 64;
  v8f acc0[4], acc1[4]; zero4(acc0); zero4(acc1);
  gemm_32x64(A, cD, W, cD, cD, m0, n0, lane, acc0, acc1);
  store_f16_16(acc0, qkv, 3 * cD, m0,      n0, lane);
  store_f16_16(acc1, qkv, 3 * cD, m0 + 16, n0, lane);
}

// Fused GEGLU FF1: h = (xn@W1a^T + b_a) * gelu(xn@W1g^T + b_g), f16 out (cM, cFF).
// Keeps 16 rows per wave: hx + gate already use 8 accumulators.
__global__ __launch_bounds__(256) void k_gemm_ff1(const _Float16* __restrict__ A,
                                                  const _Float16* __restrict__ W1,
                                                  const float* __restrict__ bias,
                                                  _Float16* __restrict__ hout) {
  int wave = (blockIdx.x * blockDim.x + threadIdx.x) >> 5;
  int lane = threadIdx.x & 31;
  const int MT = cM / 16;
  int mt = wave % MT, nt = wave / MT;
  if (nt >= cFF / 64) return;
  int m0 = mt * 16, n0 = nt * 64;
  v8f ax[4], ag[4]; zero4(ax); zero4(ag);
  for (int k0 = 0; k0 < cD; k0 += 32) {
    v16h a = load_a_frag(A, cD, m0, k0, lane);
#pragma unroll
    for (int j = 0; j < 4; ++j) {
      v16h bx = load_b_frag(W1, cD, n0 + 16 * j, k0, lane);
      ax[j] = wmma_f16(a, bx, ax[j]);
      v16h bg = load_b_frag(W1, cD, cFF + n0 + 16 * j, k0, lane);
      ag[j] = wmma_f16(a, bg, ag[j]);
    }
  }
  int hi = lane >> 4, nc = lane & 15;
#pragma unroll
  for (int j = 0; j < 4; ++j) {
    int n = n0 + 16 * j + nc;
#pragma unroll
    for (int r = 0; r < 8; ++r) {
      float hx = ax[j][r] + bias[n];
      float g  = ag[j][r] + bias[cFF + n];
      float ge = 0.5f * g * (1.f + erff(g * 0.70710678118654752f)); // exact GELU
      hout[(size_t)(m0 + r + 8 * hi) * cFF + n] = (_Float16)(hx * ge);
    }
  }
}

// ---------------- attention prep: head RMS + rotary, V transpose ----------------
// One wave per (b,h,n). Lane l owns dims l and l+32 (rotary pair).
__global__ __launch_bounds__(256) void k_qk_prep(const _Float16* __restrict__ qkv,
                                                 const float* __restrict__ theta,
                                                 const float* __restrict__ sq,
                                                 const float* __restrict__ sk,
                                                 _Float16* __restrict__ qh,
                                                 _Float16* __restrict__ kh,
                                                 _Float16* __restrict__ vt) {
  int wave = (blockIdx.x * blockDim.x + threadIdx.x) >> 5;
  int lane = threadIdx.x & 31;
  if (wave >= cB * cH * cN) return;
  int n = wave % cN, bh = wave / cN;
  int h = bh % cH, b = bh / cH;
  int m = b * cN + n;
  const _Float16* row = qkv + (size_t)m * (3 * cD);
  float th = theta[((size_t)bh * cN + n) * (cHD / 2) + lane];
  float ct = cosf(th), st = sinf(th);
#pragma unroll
  for (int which = 0; which < 2; ++which) {           // 0 = q, 1 = k
    const _Float16* src = row + which * cD + h * cHD;
    const float* sc = which ? sk : sq;
    float x1 = (float)src[lane];
    float x2 = (float)src[lane + 32];
    float ss = x1 * x1 + x2 * x2;
#pragma unroll
    for (int o = 16; o > 0; o >>= 1) ss += __shfl_xor(ss, o, 32);
    float inv = rsqrtf(ss * (1.f / cHD) + cEPS);
    x1 *= sc[lane] * inv;
    x2 *= sc[lane + 32] * inv;
    float y1 = x1 * ct - x2 * st;
    float y2 = x2 * ct + x1 * st;
    _Float16* dst = (which ? kh : qh) + ((size_t)bh * cN + n) * cHD;
    dst[lane]      = (_Float16)y1;
    dst[lane + 32] = (_Float16)y2;
  }
  // V transposed to (B,H,HD,N) so P@V B-fragments are contiguous.
  const _Float16* vsrc = row + 2 * cD + h * cHD;
  vt[((size_t)bh * cHD + lane) * cN + n]      = vsrc[lane];
  vt[((size_t)bh * cHD + lane + 32) * cN + n] = vsrc[lane + 32];
}

// ---------------- flash attention ----------------
// One wave per (b,h, 16-query tile); streams keys in tiles of 32.
__global__ __launch_bounds__(256) void k_attn(const _Float16* __restrict__ qh,
                                              const _Float16* __restrict__ kh,
                                              const _Float16* __restrict__ vt,
                                              _Float16* __restrict__ oh) {
  __shared__ _Float16 pbuf[8][16 * 32];   // per-wave P tile (C-layout -> A-layout bounce)
  int widx = threadIdx.x >> 5, lane = threadIdx.x & 31;
  int wave = blockIdx.x * 8 + widx;
  const int QT = cN / 16;
  int qt = wave % QT, bh = wave / QT;
  if (bh >= cB * cH) return;
  int m0 = qt * 16;
  const _Float16* Q = qh + (size_t)bh * cN * cHD;
  const _Float16* K = kh + (size_t)bh * cN * cHD;
  const _Float16* V = vt + (size_t)bh * cHD * cN;   // (HD, N)
  v16h aq0 = load_a_frag(Q, cHD, m0, 0, lane);
  v16h aq1 = load_a_frag(Q, cHD, m0, 32, lane);
  v8f oacc[4]; zero4(oacc);
  float mrow[8], lrow[8], alv[8];
#pragma unroll
  for (int r = 0; r < 8; ++r) { mrow[r] = -1e30f; lrow[r] = 0.f; }
  int hi = lane >> 4, nc = lane & 15;
  _Float16* pb = pbuf[widx];
  for (int kb = 0; kb < cN; kb += 32) {
    v8f s0, s1;
#pragma unroll
    for (int r = 0; r < 8; ++r) { s0[r] = 0.f; s1[r] = 0.f; }
    // S = Q @ K^T (no 1/sqrt(d): reference applies none)
    s0 = wmma_f16(aq0, load_b_frag(K, cHD, kb, 0, lane), s0);
    s0 = wmma_f16(aq1, load_b_frag(K, cHD, kb, 32, lane), s0);
    s1 = wmma_f16(aq0, load_b_frag(K, cHD, kb + 16, 0, lane), s1);
    s1 = wmma_f16(aq1, load_b_frag(K, cHD, kb + 16, 32, lane), s1);
    // online softmax; row r lives in lanes {hi*16 .. hi*16+15}
#pragma unroll
    for (int r = 0; r < 8; ++r) {
      float mx = fmaxf(s0[r], s1[r]);
#pragma unroll
      for (int d2 = 1; d2 < 16; d2 <<= 1) mx = fmaxf(mx, __shfl_xor(mx, d2, 32));
      float mn = fmaxf(mrow[r], mx);
      float p0 = __expf(s0[r] - mn), p1 = __expf(s1[r] - mn);
      float ps = p0 + p1;
#pragma unroll
      for (int d2 = 1; d2 < 16; d2 <<= 1) ps += __shfl_xor(ps, d2, 32);
      alv[r]  = __expf(mrow[r] - mn);
      lrow[r] = lrow[r] * alv[r] + ps;
      mrow[r] = mn;
      s0[r] = p0; s1[r] = p1;
    }
#pragma unroll
    for (int j = 0; j < 4; ++j)
#pragma unroll
      for (int r = 0; r < 8; ++r) oacc[j][r] *= alv[r];
    // C-layout P -> LDS (row-major 16x32) -> A-layout fragment.
    // DS ops from the same wave are in-order (no barrier needed).
#pragma unroll
    for (int r = 0; r < 8; ++r) {
      pb[(r + 8 * hi) * 32 + nc]      = (_Float16)s0[r];
      pb[(r + 8 * hi) * 32 + 16 + nc] = (_Float16)s1[r];
    }
    v16h pa = load_a_frag(pb, 32, 0, 0, lane);
#pragma unroll
    for (int j = 0; j < 4; ++j)
      oacc[j] = wmma_f16(pa, load_b_frag(V, cN, j * 16, kb, lane), oacc[j]);
  }
  int b = bh / cH, h = bh % cH;
#pragma unroll
  for (int j = 0; j < 4; ++j) {
#pragma unroll
    for (int r = 0; r < 8; ++r) {
      int m = b * cN + m0 + r + 8 * hi;
      int col = h * cHD + j * 16 + nc;
      oh[(size_t)m * cD + col] = (_Float16)(oacc[j][r] / lrow[r]);
    }
  }
}

// ---------------- host launch ----------------
extern "C" void kernel_launch(void* const* d_in, const int* in_sizes, int n_in,
                              void* d_out, int out_size, void* d_ws, size_t ws_size,
                              hipStream_t stream) {
  const float* x      = (const float*)d_in[0];
  const float* theta  = (const float*)d_in[1];
  const float* fmaps  = (const float*)d_in[2];
  const float* coords = (const float*)d_in[3];
  const float* w_corr = (const float*)d_in[4];
  const float* n1s    = (const float*)d_in[5];
  const float* sq     = (const float*)d_in[6];
  const float* sk     = (const float*)d_in[7];
  const float* w_qkv  = (const float*)d_in[8];
  const float* w_out  = (const float*)d_in[9];
  const float* n2s    = (const float*)d_in[10];
  const float* w_ff1  = (const float*)d_in[11];
  const float* b_ff1  = (const float*)d_in[12];
  const float* w_ff2  = (const float*)d_in[13];
  float* out = (float*)d_out;

  char* ws = (char*)d_ws;
  size_t off = 0;
  auto carve = [&](size_t bytes) -> char* {
    char* p = ws + off;
    off += (bytes + 255) & ~(size_t)255;
    return p;
  };
  _Float16* wcorr_h = (_Float16*)carve((size_t)cD * cC2 * 2);
  _Float16* wqkv_h  = (_Float16*)carve((size_t)3 * cD * cD * 2);
  _Float16* wout_h  = (_Float16*)carve((size_t)cD * cD * 2);
  _Float16* wff1_h  = (_Float16*)carve((size_t)2 * cFF * cD * 2);
  _Float16* wff2_h  = (_Float16*)carve((size_t)cD * cFF * 2);
  _Float16* feats_h = (_Float16*)carve((size_t)cM * cC2 * 2);
  float*    x1_f    = (float*)   carve((size_t)cM * cD * 4);
  _Float16* xn_h    = (_Float16*)carve((size_t)cM * cD * 2);
  _Float16* qkv_h   = (_Float16*)carve((size_t)cM * 3 * cD * 2);
  _Float16* q_h     = (_Float16*)carve((size_t)cM * cD * 2);
  _Float16* k_h     = (_Float16*)carve((size_t)cM * cD * 2);
  _Float16* vt_h    = (_Float16*)carve((size_t)cM * cD * 2);
  _Float16* o_h     = (_Float16*)carve((size_t)cM * cD * 2);
  float*    x2_f    = (float*)   carve((size_t)cM * cD * 4);
  _Float16* xn2_h   = (_Float16*)carve((size_t)cM * cD * 2);
  _Float16* hff_h   = (_Float16*)carve((size_t)cM * cFF * 2);

  auto cvt = [&](const float* src, _Float16* dst, int n) {
    k_cvt_f16<<<(n + 255) / 256, 256, 0, stream>>>(src, dst, n);
  };
  cvt(w_corr, wcorr_h, cD * cC2);
  cvt(w_qkv,  wqkv_h,  3 * cD * cD);
  cvt(w_out,  wout_h,  cD * cD);
  cvt(w_ff1,  wff1_h,  2 * cFF * cD);
  cvt(w_ff2,  wff2_h,  cD * cFF);

  // 1) bilinear gather
  k_feats<<<(cM * 32) / 256, 256, 0, stream>>>(fmaps, coords, feats_h);
  // 2) x1 = x + feats @ w_corr^T   (waves = 256*8)
  k_gemm_resid<<<(cM / 32) * (cD / 64) / 8, 256, 0, stream>>>(feats_h, wcorr_h, x, x1_f, cC2);
  // 3) RMS norm 1
  k_rms<<<cM / 8, 256, 0, stream>>>(x1_f, n1s, xn_h, cM, cD);
  // 4) QKV projection (waves = 256*24)
  k_gemm_qkv<<<(cM / 32) * ((3 * cD) / 64) / 8, 256, 0, stream>>>(xn_h, wqkv_h, qkv_h);
  // 5) per-head RMS + rotary + V transpose (waves = B*H*N)
  k_qk_prep<<<(cB * cH * cN) / 8, 256, 0, stream>>>(qkv_h, theta, sq, sk, q_h, k_h, vt_h);
  // 6) flash attention (waves = B*H*(N/16))
  k_attn<<<(cB * cH * (cN / 16)) / 8, 256, 0, stream>>>(q_h, k_h, vt_h, o_h);
  // 7) x2 = x1 + o @ w_out^T
  k_gemm_resid<<<(cM / 32) * (cD / 64) / 8, 256, 0, stream>>>(o_h, wout_h, x1_f, x2_f, cD);
  // 8) RMS norm 2
  k_rms<<<cM / 8, 256, 0, stream>>>(x2_f, n2s, xn2_h, cM, cD);
  // 9) fused GEGLU FF1 (waves = 512*16)
  k_gemm_ff1<<<(cM / 16) * (cFF / 64) / 8, 256, 0, stream>>>(xn2_h, wff1_h, b_ff1, hff_h);
  // 10) out = x2 + h @ w_ff2^T
  k_gemm_resid<<<(cM / 32) * (cD / 64) / 8, 256, 0, stream>>>(hff_h, wff2_h, x2_f, out, cFF);
}
